// LoRACrossAttnProcessor_54546084659992
// MI455X (gfx1250) — compile-verified, matrix operand order checked by reading, not run on hardware
//
#include <hip/hip_runtime.h>
#include <hip/hip_bf16.h>
#include <stdint.h>

// ---------------------------------------------------------------------------
// LoRA cross-attention for MI455X (gfx1250, wave32, WMMA bf16 path).
// LoRA folded exactly into base weights: W_eff = W + up @ down.
// Heavy math via v_wmma_f32_16x16x32_bf16; global->LDS staging via
// GLOBAL_LOAD_ASYNC_TO_LDS_B128 (ASYNCcnt) when the toolchain exposes it.
// ---------------------------------------------------------------------------

typedef __attribute__((ext_vector_type(16))) __bf16 v16bf;
typedef __attribute__((ext_vector_type(8)))  __bf16 v8bf;
typedef __attribute__((ext_vector_type(8)))  float  v8f;
typedef int v4i_ __attribute__((vector_size(16)));   // matches builtin param

#define HS_  1280
#define CAD_ 768
#define B_   8
#define S_   4096
#define NH_  8
#define DH_  160
#define T_   77
#define TP_  80   // K/V rows padded to 80 (multiple of 16) for the projection GEMM
#define TPV_ 96   // PV K-dim padded to 96 (multiple of 32) for WMMA

#if __has_builtin(__builtin_amdgcn_global_load_async_to_lds_b128) && \
    __has_builtin(__builtin_amdgcn_s_wait_asynccnt)
#define USE_ASYNC 1
#else
#define USE_ASYNC 0
#endif

#define AS1_ __attribute__((address_space(1)))
#define AS3_ __attribute__((address_space(3)))

// 16-byte global -> LDS copy. Async path writes LDS directly from the memory
// system (no VGPR round trip), tracked by ASYNCcnt.
__device__ __forceinline__ void g2l_b128(const unsigned short* g,
                                         unsigned short* l) {
#if USE_ASYNC
  __builtin_amdgcn_global_load_async_to_lds_b128(
      (AS1_ v4i_*)(AS1_ void*)(void*)const_cast<unsigned short*>(g),
      (AS3_ v4i_*)(AS3_ void*)(void*)l,
      /*offset=*/0, /*cpol=*/0);
#else
  *reinterpret_cast<uint4*>(l) = *reinterpret_cast<const uint4*>(g);
#endif
}

__device__ __forceinline__ void wait_async0() {
#if USE_ASYNC
  __builtin_amdgcn_s_wait_asynccnt(0);
#endif
}

__device__ __forceinline__ unsigned short f32_to_bf16(float f) {
  unsigned u = __float_as_uint(f);
  u += 0x7FFFu + ((u >> 16) & 1u);          // round-to-nearest-even
  return (unsigned short)(u >> 16);
}

// Assemble a 16x32 bf16 A/B fragment for one lane.
// ISA layout: element i -> k = (i<8 ? 0 : 16) + (lane>>4)*8 + (i&7).
// Caller passes p = &row_base[(lane>>4)*8]; lo = p[0..7], hi = p[16..23].
__device__ __forceinline__ v16bf frag_ld(const unsigned short* p) {
  v8bf lo = *reinterpret_cast<const v8bf*>(p);
  v8bf hi = *reinterpret_cast<const v8bf*>(p + 16);
  v16bf r;
#pragma unroll
  for (int i = 0; i < 8; ++i) { r[i] = lo[i]; r[i + 8] = hi[i]; }
  return r;
}

__device__ __forceinline__ v8f wmma_bf16(v16bf a, v16bf b, v8f c) {
  return __builtin_amdgcn_wmma_f32_16x16x32_bf16(
      /*neg_a=*/false, a, /*neg_b=*/false, b,
      /*c_mod=*/(short)0, c, /*reuse_a=*/false, /*reuse_b=*/false);
}

// ---------------------------------------------------------------------------
// Elementwise prep kernels
// ---------------------------------------------------------------------------
__global__ void k_cast_bf16(const float* __restrict__ x,
                            unsigned short* __restrict__ y, int n) {
  int i = blockIdx.x * 256 + threadIdx.x;
  if (i < n) y[i] = f32_to_bf16(x[i]);
}

// encoder_hidden_states [B,77,CAD] -> bf16 [B,80,CAD], rows 77..79 zeroed.
__global__ void k_pad_e(const float* __restrict__ e,
                        unsigned short* __restrict__ y) {
  int i = blockIdx.x * 256 + threadIdx.x;
  const int per_b = TP_ * CAD_;
  int b = i / per_b, rem = i % per_b, t = rem / CAD_, c = rem % CAD_;
  unsigned short v = 0;
  if (t < T_) v = f32_to_bf16(e[((size_t)b * T_ + t) * CAD_ + c]);
  y[i] = v;
}

// W_eff[n,k] = bf16( W[n,k] + sum_r up[n,r]*down[r,k] )   (rank 4)
__global__ void k_fuse_w(const float* __restrict__ W,
                         const float* __restrict__ up,
                         const float* __restrict__ down,
                         unsigned short* __restrict__ out, int N, int K) {
  int i = blockIdx.x * 256 + threadIdx.x;
  if (i >= N * K) return;
  int n = i / K, k = i % K;
  float acc = W[i];
#pragma unroll
  for (int r = 0; r < 4; ++r) acc += up[n * 4 + r] * down[r * K + k];
  out[i] = f32_to_bf16(acc);
}

// ---------------------------------------------------------------------------
// Tiled GEMM: C[M,N] = A[M,K] @ W[N,K]^T  (A,W bf16; f32 accumulate)
// Workgroup tile 128x64, 8 waves in 4x2, each wave a 32x32 tile (2x2 WMMA).
// Double-buffered LDS staging via async global->LDS copies: tile t+1 streams
// in (ASYNCcnt) while WMMAs consume tile t; one s_wait_asynccnt + barrier per
// K-step. out: bf16 (f32out==0) or f32 + bias (f32out==1).
// M%128==0, N%64==0, K%32==0 for all call sites here.
// ---------------------------------------------------------------------------
#define TM_ 128
#define TN_ 64
#define TK_ 32

__global__ __launch_bounds__(256) void k_gemm(
    const unsigned short* __restrict__ A, const unsigned short* __restrict__ W,
    const float* __restrict__ bias, unsigned short* __restrict__ outb,
    float* __restrict__ outf, int M, int N, int K, int f32out) {
  __shared__ __attribute__((aligned(16))) unsigned short lA[2][TM_ * TK_]; // 16 KB
  __shared__ __attribute__((aligned(16))) unsigned short lB[2][TN_ * TK_]; //  8 KB

  const int tid = threadIdx.x;
  const int lane = tid & 31, wave = tid >> 5;
  const int hf = lane >> 4, l = lane & 15;
  const int m0 = blockIdx.y * TM_;
  const int n0 = blockIdx.x * TN_;
  const int wm = (wave >> 1) * 32;   // 0,32,64,96
  const int wn = (wave & 1) * 32;    // 0,32

  // Per-thread staging coordinates (A: 2 chunks of 16B; B: 1 chunk).
  const int ar0 = tid >> 2,          ac0 = (tid & 3) * 8;
  const int ar1 = (tid + 256) >> 2,  ac1 = ((tid + 256) & 3) * 8;

  auto stage = [&](int buf, int k0) {
    g2l_b128(&A[(size_t)(m0 + ar0) * K + k0 + ac0], &lA[buf][ar0 * TK_ + ac0]);
    g2l_b128(&A[(size_t)(m0 + ar1) * K + k0 + ac1], &lA[buf][ar1 * TK_ + ac1]);
    g2l_b128(&W[(size_t)(n0 + ar0) * K + k0 + ac0], &lB[buf][ar0 * TK_ + ac0]);
  };

  const v8f vzero = {0.f, 0.f, 0.f, 0.f, 0.f, 0.f, 0.f, 0.f};
  v8f acc[2][2];
#pragma unroll
  for (int i = 0; i < 2; ++i)
#pragma unroll
    for (int j = 0; j < 2; ++j) acc[i][j] = vzero;

  stage(0, 0);
  wait_async0();
  __syncthreads();

  int cur = 0;
  for (int k0 = 0; k0 < K; k0 += TK_) {
    const int nxt = cur ^ 1;
    if (k0 + TK_ < K) stage(nxt, k0 + TK_);
    if (k0 + 2 * TK_ < K)  // keep a global_prefetch_b8 running ahead of async
      __builtin_prefetch(&A[(size_t)(m0 + ar0) * K + k0 + 2 * TK_], 0, 3);

    v16bf a0 = frag_ld(&lA[cur][(wm + 0 * 16 + l) * TK_ + hf * 8]);
    v16bf a1 = frag_ld(&lA[cur][(wm + 1 * 16 + l) * TK_ + hf * 8]);
    v16bf b0 = frag_ld(&lB[cur][(wn + 0 * 16 + l) * TK_ + hf * 8]);
    v16bf b1 = frag_ld(&lB[cur][(wn + 1 * 16 + l) * TK_ + hf * 8]);
    acc[0][0] = wmma_bf16(a0, b0, acc[0][0]);
    acc[0][1] = wmma_bf16(a0, b1, acc[0][1]);
    acc[1][0] = wmma_bf16(a1, b0, acc[1][0]);
    acc[1][1] = wmma_bf16(a1, b1, acc[1][1]);

    wait_async0();     // our async stores into lA/lB[nxt] have landed
    __syncthreads();   // everyone done reading [cur] and staging [nxt]
    cur = nxt;
  }

  // C/D layout: m = r + 8*(lane>>4), n = lane&15 (per 16x16 tile).
#pragma unroll
  for (int i = 0; i < 2; ++i)
#pragma unroll
    for (int j = 0; j < 2; ++j) {
      int n = n0 + wn + j * 16 + l;
#pragma unroll
      for (int r = 0; r < 8; ++r) {
        int m = m0 + wm + i * 16 + hf * 8 + r;
        float v = acc[i][j][r];
        if (f32out) outf[(size_t)m * N + n] = v + bias[n];
        else        outb[(size_t)m * N + n] = f32_to_bf16(v);
      }
    }
}

// ---------------------------------------------------------------------------
// Attention: grid (S/128, B*H), 256 threads. Each wave owns 16 query rows.
// scores = Q Kᵀ * 1/sqrt(160); masked softmax over t (t>=77 masked); O = P V.
// K staged [t(80) x k(160)] in LDS via async copies; V staged transposed
// [d(160) x t(96)], rows t in [80,96) zeroed so the PV K-loop runs 3 clean
// steps of 32. P (bf16) reuses the K LDS buffer after a barrier.
// ---------------------------------------------------------------------------
__global__ __launch_bounds__(256) void k_attn(
    const unsigned short* __restrict__ Qb, const unsigned short* __restrict__ Kb,
    const unsigned short* __restrict__ Vb, unsigned short* __restrict__ Ob) {
  __shared__ __attribute__((aligned(16))) unsigned short KP[TP_ * DH_];  // 25.6 KB
  __shared__ __attribute__((aligned(16))) unsigned short Vt[DH_ * TPV_]; // 30.7 KB

  const int bh = blockIdx.y, b = bh >> 3, hh = bh & 7;
  const int s0 = blockIdx.x * 128;
  const int tid = threadIdx.x, lane = tid & 31, wave = tid >> 5;
  const int hf = lane >> 4, l = lane & 15;

  // Stage K rows for this (b, head): contiguous 16B chunks, async into LDS.
  for (int chunk = tid; chunk < TP_ * DH_ / 8; chunk += 256) {
    int t = chunk / (DH_ / 8), co = (chunk % (DH_ / 8)) * 8;
    g2l_b128(&Kb[((size_t)(b * TP_ + t)) * HS_ + hh * DH_ + co],
             &KP[t * DH_ + co]);
  }
  // Stage Vᵀ with zero padding t in [80,96) (scalar transpose, ds stores).
  for (int idx = tid; idx < DH_ * TPV_; idx += 256) {
    int d = idx / TPV_, t = idx % TPV_;
    unsigned short v = 0;
    if (t < TP_) v = Vb[((size_t)(b * TP_ + t)) * HS_ + hh * DH_ + d];
    Vt[d * TPV_ + t] = v;
  }
  wait_async0();
  __syncthreads();

  const int qrow = s0 + wave * 16;
  const v8f vzero = {0.f, 0.f, 0.f, 0.f, 0.f, 0.f, 0.f, 0.f};
  v8f sc[5];
#pragma unroll
  for (int j = 0; j < 5; ++j) sc[j] = vzero;

  // scores[16 x 80] = Q[16 x 160] @ K[80 x 160]^T
  for (int kk = 0; kk < DH_; kk += 32) {
    v16bf a = frag_ld(
        &Qb[((size_t)(b * S_ + qrow + l)) * HS_ + hh * DH_ + kk + hf * 8]);
#pragma unroll
    for (int j = 0; j < 5; ++j) {
      v16bf bf = frag_ld(&KP[(j * 16 + l) * DH_ + kk + hf * 8]);
      sc[j] = wmma_bf16(a, bf, sc[j]);
    }
  }

  // Masked softmax per row m = r + 8*hf (row lives across 16 lanes of a half).
  const float scale = 0.0790569415f;  // 1/sqrt(160)
#pragma unroll
  for (int r = 0; r < 8; ++r) {
    float mx = -3.0e38f;
#pragma unroll
    for (int j = 0; j < 5; ++j) {
      float v = sc[j][r] * scale;
      sc[j][r] = v;
      if (j * 16 + l < T_) mx = fmaxf(mx, v);
    }
#pragma unroll
    for (int o = 8; o >= 1; o >>= 1) mx = fmaxf(mx, __shfl_xor(mx, o, 32));
    float sum = 0.f;
#pragma unroll
    for (int j = 0; j < 5; ++j) {
      float p = (j * 16 + l < T_) ? __expf(sc[j][r] - mx) : 0.f;
      sc[j][r] = p;
      sum += p;
    }
#pragma unroll
    for (int o = 8; o >= 1; o >>= 1) sum += __shfl_xor(sum, o, 32);
    float inv = 1.f / sum;
#pragma unroll
    for (int j = 0; j < 5; ++j) sc[j][r] *= inv;
  }

  __syncthreads();  // everyone done reading K before P overwrites it

  // P (bf16) into per-wave slice of KP: [16 x 96], cols 80..95 zero.
  unsigned short* Pw = &KP[wave * 16 * TPV_];
#pragma unroll
  for (int r = 0; r < 8; ++r)
#pragma unroll
    for (int j = 0; j < 5; ++j)
      Pw[(r + 8 * hf) * TPV_ + j * 16 + l] = f32_to_bf16(sc[j][r]);
  for (int e = lane; e < 16 * 16; e += 32)
    Pw[(e >> 4) * TPV_ + 80 + (e & 15)] = 0;

  // O[16 x 160] = P[16 x 96] @ Vt[160 x 96]^T  (3 K-steps of 32)
  v8f oacc[10];
#pragma unroll
  for (int j = 0; j < 10; ++j) oacc[j] = vzero;
  for (int kk = 0; kk < TPV_; kk += 32) {
    v16bf a = frag_ld(&Pw[l * TPV_ + kk + hf * 8]);
#pragma unroll
    for (int j = 0; j < 10; ++j) {
      v16bf bf = frag_ld(&Vt[(j * 16 + l) * TPV_ + kk + hf * 8]);
      oacc[j] = wmma_bf16(a, bf, oacc[j]);
    }
  }

#pragma unroll
  for (int j = 0; j < 10; ++j)
#pragma unroll
    for (int r = 0; r < 8; ++r)
      Ob[((size_t)(b * S_ + qrow + r + 8 * hf)) * HS_ + hh * DH_ + j * 16 + l] =
          f32_to_bf16(oacc[j][r]);
}

// ---------------------------------------------------------------------------
// Host launcher
// ---------------------------------------------------------------------------
extern "C" void kernel_launch(void* const* d_in, const int* in_sizes, int n_in,
                              void* d_out, int out_size, void* d_ws,
                              size_t ws_size, hipStream_t stream) {
  (void)in_sizes; (void)n_in; (void)out_size; (void)ws_size;
  const float* hs     = (const float*)d_in[0];
  const float* ehs    = (const float*)d_in[1];
  const float* Wq     = (const float*)d_in[2];
  const float* Wk     = (const float*)d_in[3];
  const float* Wv     = (const float*)d_in[4];
  const float* Wo     = (const float*)d_in[5];
  const float* bo     = (const float*)d_in[6];
  const float* q_down = (const float*)d_in[7];
  const float* q_up   = (const float*)d_in[8];
  const float* k_down = (const float*)d_in[9];
  const float* k_up   = (const float*)d_in[10];
  const float* v_down = (const float*)d_in[11];
  const float* v_up   = (const float*)d_in[12];
  const float* o_down = (const float*)d_in[13];
  const float* o_up   = (const float*)d_in[14];
  float* out = (float*)d_out;

  // Workspace layout (bytes, 256-aligned). O reuses the X buffer (X is dead
  // after the Q projection).
  const size_t SZ_X  = (size_t)B_ * S_ * HS_ * 2;     // 83,886,080
  const size_t SZ_E  = (size_t)B_ * TP_ * CAD_ * 2;   //    983,040
  const size_t SZ_WQ = (size_t)HS_ * HS_ * 2;         //  3,276,800
  const size_t SZ_WK = (size_t)HS_ * CAD_ * 2;        //  1,966,080
  const size_t SZ_KV = (size_t)B_ * TP_ * HS_ * 2;    //  1,638,400
  char* ws = (char*)d_ws;
  unsigned short* Xb  = (unsigned short*)(ws);                 // X, later O
  unsigned short* Eb  = (unsigned short*)(ws + SZ_X);
  unsigned short* Wqe = (unsigned short*)(ws + SZ_X + SZ_E);
  unsigned short* Wke = (unsigned short*)(ws + SZ_X + SZ_E + SZ_WQ);
  unsigned short* Wve = (unsigned short*)(ws + SZ_X + SZ_E + SZ_WQ + SZ_WK);
  unsigned short* Woe = (unsigned short*)(ws + SZ_X + SZ_E + SZ_WQ + 2 * SZ_WK);
  unsigned short* Qb  = (unsigned short*)(ws + SZ_X + SZ_E + 2 * SZ_WQ + 2 * SZ_WK);
  unsigned short* Kb  = (unsigned short*)((char*)Qb + SZ_X);
  unsigned short* Vb  = (unsigned short*)((char*)Kb + SZ_KV);
  unsigned short* Ob  = Xb;

  // 1) casts
  {
    int n = B_ * S_ * HS_;
    k_cast_bf16<<<n / 256, 256, 0, stream>>>(hs, Xb, n);
  }
  k_pad_e<<<(B_ * TP_ * CAD_) / 256, 256, 0, stream>>>(ehs, Eb);

  // 2) fold LoRA into weights
  k_fuse_w<<<(HS_ * HS_) / 256, 256, 0, stream>>>(Wq, q_up, q_down, Wqe, HS_, HS_);
  k_fuse_w<<<(HS_ * CAD_) / 256, 256, 0, stream>>>(Wk, k_up, k_down, Wke, HS_, CAD_);
  k_fuse_w<<<(HS_ * CAD_) / 256, 256, 0, stream>>>(Wv, v_up, v_down, Wve, HS_, CAD_);
  k_fuse_w<<<(HS_ * HS_) / 256, 256, 0, stream>>>(Wo, o_up, o_down, Woe, HS_, HS_);

  // 3) projections (WMMA GEMMs)
  k_gemm<<<dim3(HS_ / TN_, (B_ * S_) / TM_), 256, 0, stream>>>(
      Xb, Wqe, nullptr, Qb, nullptr, B_ * S_, HS_, HS_, 0);
  k_gemm<<<dim3(HS_ / TN_, (B_ * TP_) / TM_), 256, 0, stream>>>(
      Eb, Wke, nullptr, Kb, nullptr, B_ * TP_, HS_, CAD_, 0);
  k_gemm<<<dim3(HS_ / TN_, (B_ * TP_) / TM_), 256, 0, stream>>>(
      Eb, Wve, nullptr, Vb, nullptr, B_ * TP_, HS_, CAD_, 0);

  // 4) attention (WMMA QKᵀ + softmax + WMMA PV)
  k_attn<<<dim3(S_ / 128, B_ * NH_), 256, 0, stream>>>(Qb, Kb, Vb, Ob);

  // 5) output projection + bias -> f32
  k_gemm<<<dim3(HS_ / TN_, (B_ * S_) / TM_), 256, 0, stream>>>(
      Ob, Woe, bo, nullptr, out, B_ * S_, HS_, HS_, 1);
}
